// LSTM_65231963292326
// MI455X (gfx1250) — matrix operand order repaired and checked by reading, not compiled, not run on hardware
//
#include <hip/hip_runtime.h>

// ---------------------------------------------------------------------------
// Problem constants (fixed by the reference)
// ---------------------------------------------------------------------------
#define EDIM   300
#define EPAD   320        // pad K of layer0 to a multiple of 32
#define HDIM   512
#define BSZ    64
#define TLEN   512
#define G4H    2048       // 4*H
#define MROWS  (BSZ*TLEN) // 32768

typedef __attribute__((ext_vector_type(16))) __bf16       v16bf;
typedef __attribute__((ext_vector_type(8)))  float        v8f;
typedef __attribute__((ext_vector_type(4)))  unsigned int u32x4;

union FragBF { u32x4 u[2]; v16bf v; };

__device__ __forceinline__ unsigned short f2bf(float f) {
    unsigned u = __float_as_uint(f);
    unsigned r = u + 0x7FFFu + ((u >> 16) & 1u);
    return (unsigned short)(r >> 16);
}
__device__ __forceinline__ float sigm(float x) {
    return 1.0f / (1.0f + __expf(-x));
}

// CDNA5 async global->LDS copy of one 16B chunk per lane (ASYNCcnt-tracked).
// lds_off = wave-relative LDS byte offset (low 32 bits of flat shared ptr).
__device__ __forceinline__ void async_g2l_b128(unsigned lds_off, unsigned long long gaddr) {
    asm volatile("global_load_async_to_lds_b128 %0, %1, off"
                 :: "v"(lds_off), "v"(gaddr) : "memory");
}
__device__ __forceinline__ void wait_asynccnt0() {
#if __has_builtin(__builtin_amdgcn_s_wait_asynccnt)
    __builtin_amdgcn_s_wait_asynccnt(0);
#else
    asm volatile("s_wait_asynccnt 0x0" ::: "memory");
#endif
}

// ---------------------------------------------------------------------------
// f32 -> bf16 convert with K padding (rows x ks -> rows x kd, zero-padded)
// ---------------------------------------------------------------------------
__global__ void cvt_pad_bf16(const float* __restrict__ src,
                             unsigned short* __restrict__ dst,
                             int rows, int ks, int kd) {
    int i = blockIdx.x * blockDim.x + threadIdx.x;
    int total = rows * kd;
    if (i >= total) return;
    int r = i / kd, k = i - r * kd;
    float v = (k < ks) ? src[(size_t)r * ks + k] : 0.0f;
    dst[i] = f2bf(v);
}

__global__ void zero_u32(unsigned* __restrict__ p, int n) {
    int i = blockIdx.x * blockDim.x + threadIdx.x;
    if (i < n) p[i] = 0u;
}

// ---------------------------------------------------------------------------
// WMMA GEMM:  out[t][b][n] = A[b*T+t][:] . W[n][:] + bias[n]
//   A: [32768, K] bf16 row-major, W: [2048, K] bf16 row-major (i.e. X @ W^T)
//   out: f32 [T, B, 2048] (time-major so the scan reads a contiguous slice)
// Block = 256 threads = 8 waves; block tile 128x128; wave tile 32x64.
// ---------------------------------------------------------------------------
__global__ void wmma_gemm_xg(const unsigned short* __restrict__ A,
                             const unsigned short* __restrict__ W,
                             const float* __restrict__ bias,
                             float* __restrict__ out,
                             int K) {
    const int lane = threadIdx.x & 31;
    const int wv   = threadIdx.x >> 5;
    const int wrow = wv & 3;           // 4 wave rows
    const int wcol = wv >> 2;          // 2 wave cols
    const int row0 = blockIdx.y * 128 + wrow * 32;
    const int col0 = blockIdx.x * 128 + wcol * 64;
    const int lr   = lane & 15;
    const int hf   = lane >> 4;

    v8f acc[2][4];
#pragma unroll
    for (int i = 0; i < 2; ++i)
#pragma unroll
        for (int j = 0; j < 4; ++j)
            acc[i][j] = (v8f){0.f,0.f,0.f,0.f,0.f,0.f,0.f,0.f};

    for (int kc = 0; kc < K; kc += 32) {
        FragBF a[2];
#pragma unroll
        for (int i = 0; i < 2; ++i) {
            const unsigned short* p = A + (size_t)(row0 + 16*i + lr) * K + kc + hf*8;
            a[i].u[0] = *(const u32x4*)p;
            a[i].u[1] = *(const u32x4*)(p + 16);
        }
        FragBF bq[4];
#pragma unroll
        for (int j = 0; j < 4; ++j) {
            const unsigned short* p = W + (size_t)(col0 + 16*j + lr) * K + kc + hf*8;
            bq[j].u[0] = *(const u32x4*)p;
            bq[j].u[1] = *(const u32x4*)(p + 16);
        }
#pragma unroll
        for (int i = 0; i < 2; ++i)
#pragma unroll
            for (int j = 0; j < 4; ++j)
                acc[i][j] = __builtin_amdgcn_wmma_f32_16x16x32_bf16(
                    false, a[i].v, false, bq[j].v, (short)0, acc[i][j], false, false);
    }

#pragma unroll
    for (int i = 0; i < 2; ++i) {
        int mbase = row0 + 16*i + hf*8;
#pragma unroll
        for (int j = 0; j < 4; ++j) {
            int col = col0 + 16*j + lr;
            float bv = bias[col];
#pragma unroll
            for (int r = 0; r < 8; ++r) {
                int m = mbase + r;
                int t = m & (TLEN - 1);
                int b = m >> 9;
                out[((size_t)t * BSZ + b) * G4H + col] = acc[i][j][r] + bv;
            }
        }
    }
}

// ---------------------------------------------------------------------------
// Persistent bidirectional LSTM scan.
// Grid = 32 blocks x 256 threads. dir = blockIdx.x>>4 (0 fwd, 1 bwd).
// Each WG owns 32 hidden units (j0 = wg*32) -> 128 gate columns (i,f,g,o).
// LDS (224KB dynamic): 128KB Whh slice (loaded once, async DMA),
//                      64KB h stage (async DMA per step), 32KB gate tiles.
// Per step:  async-stage h[64,512] bf16 into LDS, gates = xg[t] + h @ Whh^T
// via WMMA (A from LDS, B from LDS), gate tiles -> LDS, elementwise cell
// update (c in regs), h -> global double buffer, per-direction atomic grid
// barrier.
// ---------------------------------------------------------------------------
__global__ void lstm_scan(const float* __restrict__ xgf,
                          const float* __restrict__ xgb,
                          const unsigned short* __restrict__ whhf,
                          const unsigned short* __restrict__ whhb,
                          unsigned short* __restrict__ hping,  // [2 dirs][2 bufs][B*H] bf16
                          float* __restrict__ hfin,            // [2 dirs][B*H] f32
                          unsigned short* __restrict__ outbf,  // [B*T, 2H] bf16
                          unsigned* __restrict__ syncc,        // [2] counters
                          int writeOut) {
    extern __shared__ char smem[];
    unsigned short* whl = (unsigned short*)smem;                 // [128][512] bf16 (128KB)
    unsigned short* hsh = (unsigned short*)(smem + 128*1024);    // [64][512] bf16  (64KB)
    float*          gsh = (float*)(smem + 192*1024);             // [64][128] f32   (32KB)

    const int tid  = threadIdx.x;
    const int lane = tid & 31;
    const int wv   = tid >> 5;
    const int dir  = blockIdx.x >> 4;
    const int wg   = blockIdx.x & 15;
    const int j0   = wg * 32;

    const float*          xg  = dir ? xgb  : xgf;
    const unsigned short* whh = dir ? whhb : whhf;

    const int mtile = wv & 3;      // batch tile: rows 16*mtile..+15
    const int half  = wv >> 2;     // gate-column half: ntiles half*4..half*4+3
    const int lr    = lane & 15;
    const int hf    = lane >> 4;

    // ---- one-time async preload of this WG's Whh slice into LDS ----
    // local row r = g*32 + jj  ->  global row g*H + j0 + jj ; 64 16B chunks/row
    {
        unsigned lbase = (unsigned)(unsigned long long)(void*)whl;
        for (int i = tid; i < (128*HDIM*2)/16; i += 256) {
            int r  = i >> 6, c = i & 63;
            int g  = r >> 5, jj = r & 31;
            const unsigned short* gp = whh + (size_t)(g*HDIM + j0 + jj)*HDIM + c*8;
            async_g2l_b128(lbase + (unsigned)i*16u, (unsigned long long)gp);
        }
    }

    float creg[8];
#pragma unroll
    for (int q = 0; q < 8; ++q) creg[q] = 0.0f;

    for (int s = 0; s < TLEN; ++s) {
        const int tt = dir ? (TLEN - 1 - s) : s;
        unsigned short* cur = hping + ((size_t)dir*2 + (s & 1))       * (BSZ*HDIM);
        unsigned short* nxt = hping + ((size_t)dir*2 + ((s + 1) & 1)) * (BSZ*HDIM);

        // async-stage full h (state s) into LDS: 4096 x 16B
        {
            unsigned lbase = (unsigned)(unsigned long long)(void*)hsh;
            unsigned long long gbase = (unsigned long long)cur;
            for (int i = tid; i < (BSZ*HDIM*2)/16; i += 256)
                async_g2l_b128(lbase + (unsigned)i*16u, gbase + (unsigned long long)i*16u);
        }
        wait_asynccnt0();
        __syncthreads();

        // gates tile = h @ Whh^T for this WG's 128 gate columns (A,B from LDS)
        v8f acc[4];
#pragma unroll
        for (int j = 0; j < 4; ++j) acc[j] = (v8f){0.f,0.f,0.f,0.f,0.f,0.f,0.f,0.f};

        for (int kc = 0; kc < HDIM; kc += 32) {
            FragBF a;
            {
                const unsigned short* p = hsh + (size_t)(16*mtile + lr) * HDIM + kc + hf*8;
                a.u[0] = *(const u32x4*)p;
                a.u[1] = *(const u32x4*)(p + 16);
            }
            FragBF bq[4];
#pragma unroll
            for (int jn = 0; jn < 4; ++jn) {
                int nt   = half*4 + jn;
                int lrow = (nt >> 1)*32 + (nt & 1)*16 + lr;   // local Whh row
                const unsigned short* p = whl + (size_t)lrow * HDIM + kc + hf*8;
                bq[jn].u[0] = *(const u32x4*)p;
                bq[jn].u[1] = *(const u32x4*)(p + 16);
            }
#pragma unroll
            for (int jn = 0; jn < 4; ++jn)
                acc[jn] = __builtin_amdgcn_wmma_f32_16x16x32_bf16(
                    false, a.v, false, bq[jn].v, (short)0, acc[jn], false, false);
        }

        // spill gate tiles to LDS: row = batch, col = gate*32 + jj (jj in 0..31)
#pragma unroll
        for (int jn = 0; jn < 4; ++jn) {
            int nt  = half*4 + jn;
            int cb  = (nt >> 1)*32 + (nt & 1)*16 + lr;
#pragma unroll
            for (int r = 0; r < 8; ++r) {
                int row = 16*mtile + hf*8 + r;
                gsh[row*128 + cb] = acc[jn][r];
            }
        }
        __syncthreads();

        // elementwise LSTM cell: 2048 (b,jj) elems, 8 per thread, c in regs
        const float* xgrow = xg + (size_t)tt * BSZ * G4H;
#pragma unroll
        for (int q = 0; q < 8; ++q) {
            int idx = tid + 256*q;
            int b   = idx >> 5;
            int jj  = idx & 31;
            int jg  = j0 + jj;
            float gi = xgrow[(size_t)b*G4H +            jg] + gsh[b*128 +       jj];
            float gf = xgrow[(size_t)b*G4H + HDIM     + jg] + gsh[b*128 + 32  + jj];
            float gg = xgrow[(size_t)b*G4H + 2*HDIM   + jg] + gsh[b*128 + 64  + jj];
            float go = xgrow[(size_t)b*G4H + 3*HDIM   + jg] + gsh[b*128 + 96  + jj];
            float c  = sigm(gf)*creg[q] + sigm(gi)*tanhf(gg);
            creg[q]  = c;
            float h  = sigm(go)*tanhf(c);
            unsigned short hb = f2bf(h);
            nxt[(size_t)b*HDIM + jg] = hb;
            hfin[(size_t)dir*(BSZ*HDIM) + (size_t)b*HDIM + jg] = h;
            if (writeOut)
                outbf[((size_t)b*TLEN + tt)*(2*HDIM) + (size_t)dir*HDIM + jg] = hb;
        }

        // per-direction grid barrier (16 WGs), monotonic counter
        __syncthreads();
        if (tid == 0) {
            __threadfence();
            atomicAdd(&syncc[dir], 1u);
            unsigned target = 16u * (unsigned)(s + 1);
            while (atomicAdd(&syncc[dir], 0u) < target)
                __builtin_amdgcn_s_sleep(2);
            __threadfence();
        }
        __syncthreads();
    }
}

// ---------------------------------------------------------------------------
// Head: out[b] = concat(h_f, h_b) . w_out + b_out
// ---------------------------------------------------------------------------
__global__ void head_kernel(const float* __restrict__ hfin,
                            const float* __restrict__ wout,
                            const float* __restrict__ bout,
                            float* __restrict__ out) {
    __shared__ float red[256];
    int b = blockIdx.x;
    float s = 0.0f;
    for (int j = threadIdx.x; j < HDIM; j += 256) {
        s += hfin[(size_t)b*HDIM + j] * wout[j];
        s += hfin[(size_t)(BSZ*HDIM) + (size_t)b*HDIM + j] * wout[HDIM + j];
    }
    red[threadIdx.x] = s;
    __syncthreads();
    for (int st = 128; st > 0; st >>= 1) {
        if (threadIdx.x < st) red[threadIdx.x] += red[threadIdx.x + st];
        __syncthreads();
    }
    if (threadIdx.x == 0) out[b] = red[0] + bout[0];
}

// ---------------------------------------------------------------------------
// Host orchestration
// ---------------------------------------------------------------------------
extern "C" void kernel_launch(void* const* d_in, const int* in_sizes, int n_in,
                              void* d_out, int out_size, void* d_ws, size_t ws_size,
                              hipStream_t stream) {
    const float* text    = (const float*)d_in[0];
    const float* wih_l0f = (const float*)d_in[1];
    const float* whh_l0f = (const float*)d_in[2];
    const float* b_l0f   = (const float*)d_in[3];
    const float* wih_l0b = (const float*)d_in[4];
    const float* whh_l0b = (const float*)d_in[5];
    const float* b_l0b   = (const float*)d_in[6];
    const float* wih_l1f = (const float*)d_in[7];
    const float* whh_l1f = (const float*)d_in[8];
    const float* b_l1f   = (const float*)d_in[9];
    const float* wih_l1b = (const float*)d_in[10];
    const float* whh_l1b = (const float*)d_in[11];
    const float* b_l1b   = (const float*)d_in[12];
    const float* w_out   = (const float*)d_in[13];
    const float* b_out   = (const float*)d_in[14];
    float* out = (float*)d_out;

    // workspace carving (256B aligned)
    char* base = (char*)d_ws;
    size_t off = 0;
    auto carve = [&](size_t bytes) -> char* {
        char* p = base + off;
        off += (bytes + 255) & ~(size_t)255;
        return p;
    };
    unsigned short* xbf    = (unsigned short*)carve((size_t)MROWS * EPAD * 2);
    unsigned short* w0fbf  = (unsigned short*)carve((size_t)G4H * EPAD * 2);
    unsigned short* w0bbf  = (unsigned short*)carve((size_t)G4H * EPAD * 2);
    unsigned short* w1fbf  = (unsigned short*)carve((size_t)G4H * 1024 * 2);
    unsigned short* w1bbf  = (unsigned short*)carve((size_t)G4H * 1024 * 2);
    unsigned short* whh0f  = (unsigned short*)carve((size_t)G4H * HDIM * 2);
    unsigned short* whh0b  = (unsigned short*)carve((size_t)G4H * HDIM * 2);
    unsigned short* whh1f  = (unsigned short*)carve((size_t)G4H * HDIM * 2);
    unsigned short* whh1b  = (unsigned short*)carve((size_t)G4H * HDIM * 2);
    unsigned short* out0bf = (unsigned short*)carve((size_t)MROWS * 1024 * 2);
    float*          xgF    = (float*)carve((size_t)TLEN * BSZ * G4H * 4);
    float*          xgB    = (float*)carve((size_t)TLEN * BSZ * G4H * 4);
    unsigned short* hping  = (unsigned short*)carve((size_t)2 * 2 * BSZ * HDIM * 2);
    unsigned*       syncc  = (unsigned*)carve(256);   // contiguous right after hping
    float*          hfin   = (float*)carve((size_t)2 * BSZ * HDIM * 4);

    const int THR = 256;
    auto blocks = [&](size_t n) { return (unsigned)((n + THR - 1) / THR); };

    // --- convert weights / input to bf16 (padded) ---
    cvt_pad_bf16<<<blocks((size_t)MROWS*EPAD), THR, 0, stream>>>(text,    xbf,   MROWS, EDIM, EPAD);
    cvt_pad_bf16<<<blocks((size_t)G4H*EPAD),   THR, 0, stream>>>(wih_l0f, w0fbf, G4H,  EDIM, EPAD);
    cvt_pad_bf16<<<blocks((size_t)G4H*EPAD),   THR, 0, stream>>>(wih_l0b, w0bbf, G4H,  EDIM, EPAD);
    cvt_pad_bf16<<<blocks((size_t)G4H*1024),   THR, 0, stream>>>(wih_l1f, w1fbf, G4H, 1024, 1024);
    cvt_pad_bf16<<<blocks((size_t)G4H*1024),   THR, 0, stream>>>(wih_l1b, w1bbf, G4H, 1024, 1024);
    cvt_pad_bf16<<<blocks((size_t)G4H*HDIM),   THR, 0, stream>>>(whh_l0f, whh0f, G4H, HDIM, HDIM);
    cvt_pad_bf16<<<blocks((size_t)G4H*HDIM),   THR, 0, stream>>>(whh_l0b, whh0b, G4H, HDIM, HDIM);
    cvt_pad_bf16<<<blocks((size_t)G4H*HDIM),   THR, 0, stream>>>(whh_l1f, whh1f, G4H, HDIM, HDIM);
    cvt_pad_bf16<<<blocks((size_t)G4H*HDIM),   THR, 0, stream>>>(whh_l1b, whh1b, G4H, HDIM, HDIM);

    dim3 gemm_grid(G4H / 128, MROWS / 128);

    // --- layer 0: xg GEMMs (K=320), then scan ---
    wmma_gemm_xg<<<gemm_grid, THR, 0, stream>>>(xbf, w0fbf, b_l0f, xgF, EPAD);
    wmma_gemm_xg<<<gemm_grid, THR, 0, stream>>>(xbf, w0bbf, b_l0b, xgB, EPAD);

    int zero_n = (2*2*BSZ*HDIM*2 + 256) / 4;   // hping + sync counters
    zero_u32<<<blocks(zero_n), THR, 0, stream>>>((unsigned*)hping, zero_n);

    lstm_scan<<<32, THR, 224*1024, stream>>>(xgF, xgB, whh0f, whh0b,
                                             hping, hfin, out0bf, syncc, 1);

    // --- layer 1: xg GEMMs over concat output (K=1024), then scan ---
    wmma_gemm_xg<<<gemm_grid, THR, 0, stream>>>(out0bf, w1fbf, b_l1f, xgF, 1024);
    wmma_gemm_xg<<<gemm_grid, THR, 0, stream>>>(out0bf, w1bbf, b_l1b, xgB, 1024);

    zero_u32<<<blocks(zero_n), THR, 0, stream>>>((unsigned*)hping, zero_n);

    lstm_scan<<<32, THR, 224*1024, stream>>>(xgF, xgB, whh1f, whh1b,
                                             hping, hfin, out0bf, syncc, 0);

    // --- head ---
    head_kernel<<<BSZ, THR, 0, stream>>>(hfin, w_out, b_out, out);

    (void)in_sizes; (void)n_in; (void)out_size; (void)ws_size;
}
